// Decoder_85504208929291
// MI455X (gfx1250) — matrix operand order repaired and checked by reading, not compiled, not run on hardware
//
#include <hip/hip_runtime.h>
#include <hip/hip_bf16.h>
#include <cstddef>

typedef _Float16 h16;
typedef __attribute__((ext_vector_type(16))) _Float16 v16h;
typedef __attribute__((ext_vector_type(8)))  _Float16 v8h;
typedef __attribute__((ext_vector_type(8)))  float    v8f;
typedef __attribute__((__vector_size__(16))) int      v4i;

union V16 { v16h v; v8h h[2]; };

#define B_   64
#define L_   20
#define VN_  196
#define VD_  512
#define E_   512
#define H_   512
#define V_   32000
#define KX_  1536   // ctx(512) + emb(512) + h(512)
#define G4_  2048   // 4*H

#if defined(__has_builtin)
#  if __has_builtin(__builtin_amdgcn_global_load_async_to_lds_b128)
#    define HAVE_ASYNC_LDS 1
#  endif
#endif
#ifndef HAVE_ASYNC_LDS
#  define HAVE_ASYNC_LDS 0
#endif

typedef __attribute__((address_space(1))) v4i* gv4i_p;
typedef __attribute__((address_space(3))) v4i* lv4i_p;

__device__ __forceinline__ void wait_asynccnt0() {
#if defined(__has_builtin)
#  if __has_builtin(__builtin_amdgcn_s_wait_asynccnt)
  __builtin_amdgcn_s_wait_asynccnt(0);
  return;
#  else
  asm volatile("s_wait_asynccnt 0" ::: "memory");
  return;
#  endif
#else
  asm volatile("s_wait_asynccnt 0" ::: "memory");
#endif
}

// ---------------------------------------------------------------------------
// Weight conversion: W_cat[n][k] f16, n-major (B-operand friendly)
__global__ void k_cvt_wcat(const float* __restrict__ Wih, const float* __restrict__ Whh,
                           h16* __restrict__ Wc) {
  int idx = blockIdx.x * 256 + threadIdx.x;           // 2048*1536 elements
  int n = idx / KX_, k = idx - n * KX_;
  float v = (k < 1024) ? Wih[(size_t)n * 1024 + k] : Whh[(size_t)n * 512 + (k - 1024)];
  Wc[idx] = (h16)v;
}

// out_W (512,32000) f32  ->  outWt (32000,512) f16  (tiled transpose)
__global__ void k_cvt_outw(const float* __restrict__ W, h16* __restrict__ Wt) {
  __shared__ float tile[32][33];
  int bx = blockIdx.x, by = blockIdx.y;               // (1000, 16)
  int tx = threadIdx.x, ty = threadIdx.y;             // (32, 8)
#pragma unroll
  for (int i = 0; i < 4; i++) {
    int kk = by * 32 + ty + i * 8;
    tile[ty + i * 8][tx] = W[(size_t)kk * V_ + bx * 32 + tx];
  }
  __syncthreads();
#pragma unroll
  for (int i = 0; i < 4; i++) {
    int nn = bx * 32 + ty + i * 8;
    Wt[(size_t)nn * 512 + by * 32 + tx] = (h16)tile[tx][ty + i * 8];
  }
}

// ---------------------------------------------------------------------------
// Init: t_emb gather, feature mean, h0/c0, (time-invariant) attention -> ctx,
// and fill Xcat[b] = [ctx | emb_0 | h0] in f16.
__global__ void k_init(const float* __restrict__ features, const int* __restrict__ topics,
                       const int* __restrict__ captions,
                       const float* __restrict__ Whv, const float* __restrict__ Wcv,
                       const float* __restrict__ Wht, const float* __restrict__ Wct,
                       const float* __restrict__ vw,
                       const float* __restrict__ topic_embed, const float* __restrict__ word_embed,
                       float* __restrict__ h_state, float* __restrict__ c_state,
                       h16* __restrict__ Xcat) {
  __shared__ float s_fmean[512], s_temb[512], s_alpha[256], s_red[256];
  int b = blockIdx.x, tid = threadIdx.x;
  const float* F = features + (size_t)b * VN_ * VD_;
  int topic = topics[b];
  for (int j = tid; j < 512; j += 256) s_temb[j] = topic_embed[(size_t)topic * 512 + j];
  for (int j = tid; j < 512; j += 256) {
    float acc = 0.f;
    for (int vn = 0; vn < VN_; vn++) acc += F[vn * VD_ + j];
    s_fmean[j] = acc * (1.f / (float)VN_);
  }
  __syncthreads();
  for (int j = tid; j < 512; j += 256) {
    float ah = 0.f, ac = 0.f;
    for (int k = 0; k < 512; k++) {
      float fm = s_fmean[k], te = s_temb[k];
      ah += fm * Whv[k * 512 + j] + te * Wht[k * 512 + j];
      ac += fm * Wcv[k * 512 + j] + te * Wct[k * 512 + j];
    }
    float h0 = tanhf(ah), c0 = tanhf(ac);
    h_state[b * 512 + j] = h0;
    c_state[b * 512 + j] = c0;
    Xcat[(size_t)b * KX_ + 1024 + j] = (h16)h0;
  }
  __syncthreads();
  // attention logits (softmax shift-invariance: h-term and biases cancel)
  for (int j = tid; j < 512; j += 256) s_fmean[j] = vw[j];   // reuse as attn_vw
  __syncthreads();
  float logit = -3.0e38f;
  if (tid < VN_) {
    float s = 0.f;
    const float* fr = F + tid * VD_;
    for (int k = 0; k < 512; k++) s += fr[k] * s_fmean[k];
    logit = s;
  }
  s_red[tid] = logit;
  __syncthreads();
  for (int st = 128; st > 0; st >>= 1) {
    if (tid < st) s_red[tid] = fmaxf(s_red[tid], s_red[tid + st]);
    __syncthreads();
  }
  float mx = s_red[0];
  __syncthreads();
  float e = (tid < VN_) ? __expf(logit - mx) : 0.f;
  s_red[tid] = e;
  __syncthreads();
  for (int st = 128; st > 0; st >>= 1) {
    if (tid < st) s_red[tid] += s_red[tid + st];
    __syncthreads();
  }
  float inv = 1.f / s_red[0];
  __syncthreads();
  s_alpha[tid] = e * inv;
  __syncthreads();
  int cap0 = captions[b * L_ + 0];
  for (int j = tid; j < 512; j += 256) {
    float acc = 0.f;
    for (int vn = 0; vn < VN_; vn++) acc += s_alpha[vn] * F[vn * VD_ + j];
    Xcat[(size_t)b * KX_ + j]       = (h16)acc;                                 // ctx (const over t)
    Xcat[(size_t)b * KX_ + 512 + j] = (h16)word_embed[(size_t)cap0 * 512 + j];  // emb_0
  }
}

// ---------------------------------------------------------------------------
// Gates GEMM: G(64x2048) = Xcat(64x1536,f16) @ Wcat^T  via WMMA f16->f32
__global__ void k_gates(const h16* __restrict__ X, const h16* __restrict__ Wc,
                        float* __restrict__ G) {
  int wave = (blockIdx.x * blockDim.x + threadIdx.x) >> 5;  // 0..511
  int lane = threadIdx.x & 31;
  int mTile = wave & 3;                                     // 64/16
  int nTile = wave >> 2;                                    // 2048/16
  int row0 = mTile * 16, n0 = nTile * 16;
  int r = lane & 15, hi = lane >> 4;
  v8f acc = {};
  for (int k0 = 0; k0 < KX_; k0 += 32) {
    const h16* pa = X + (size_t)(row0 + r) * KX_ + k0 + hi * 16;
    V16 a; a.h[0] = *(const v8h*)pa; a.h[1] = *(const v8h*)(pa + 8);
    const h16* pb = Wc + (size_t)(n0 + r) * KX_ + k0 + hi * 16;
    v16h bfrag = *(const v16h*)pb;
    acc = __builtin_amdgcn_wmma_f32_16x16x32_f16(false, a.v, false, bfrag,
                                                 (short)0, acc, false, false);
  }
#pragma unroll
  for (int i = 0; i < 8; i++)
    G[(size_t)(row0 + i + hi * 8) * G4_ + n0 + r] = acc[i];
}

// ---------------------------------------------------------------------------
// LSTM pointwise; records h_t (f16) and preloads Xcat for step t+1
__global__ void k_lstm(const float* __restrict__ G, const float* __restrict__ bih,
                       const float* __restrict__ bhh,
                       float* __restrict__ h_state, float* __restrict__ c_state,
                       h16* __restrict__ Hall, h16* __restrict__ Xcat,
                       const float* __restrict__ word_embed, const int* __restrict__ captions,
                       int t) {
  int idx = blockIdx.x * blockDim.x + threadIdx.x;     // 64*512
  int b = idx >> 9, j = idx & 511;
  const float* g = G + (size_t)b * G4_;
  float i_ = g[j]        + bih[j]        + bhh[j];
  float f_ = g[512 + j]  + bih[512 + j]  + bhh[512 + j];
  float gg = g[1024 + j] + bih[1024 + j] + bhh[1024 + j];
  float o_ = g[1536 + j] + bih[1536 + j] + bhh[1536 + j];
  float si = 1.f / (1.f + __expf(-i_));
  float sf = 1.f / (1.f + __expf(-f_));
  float so = 1.f / (1.f + __expf(-o_));
  float c_new = sf * c_state[idx] + si * tanhf(gg);
  float h_new = so * tanhf(c_new);
  c_state[idx] = c_new;
  h_state[idx] = h_new;
  Hall[((size_t)b * L_ + t) * 512 + j] = (h16)h_new;
  Xcat[(size_t)b * KX_ + 1024 + j] = (h16)h_new;
  if (t + 1 < L_) {
    int cap = captions[b * L_ + t + 1];
    Xcat[(size_t)b * KX_ + 512 + j] = (h16)word_embed[(size_t)cap * 512 + j];
  }
}

// ---------------------------------------------------------------------------
// Output GEMM: logits(1280x32000) = Hall(1280x512,f16) @ outWt^T + bias
// WG = 8 waves; WG tile 64(M) x 128(N); A tile (64x512 f16 = 64KB) staged in
// LDS (async-to-LDS when available); each wave: 1 A-frag x 4 B-frags ->
// 4 independent WMMA accumulator chains per K chunk.
__global__ void k_outgemm(const h16* __restrict__ Hh, const h16* __restrict__ Bt,
                          const float* __restrict__ bias, float* __restrict__ Out) {
  __shared__ h16 At[64 * 512];
  int row0 = blockIdx.x * 64;        // 1280/64 = 20
  int nBase = blockIdx.y * 128;      // 32000/128 = 250
  int tid = threadIdx.x;
#if HAVE_ASYNC_LDS
  {
    char* src = (char*)(Hh + (size_t)row0 * 512);
    char* dst = (char*)At;
#pragma unroll
    for (int it = 0; it < 16; ++it) {
      int off = (it * 256 + tid) * 16;
      __builtin_amdgcn_global_load_async_to_lds_b128(
          (gv4i_p)(src + off), (lv4i_p)(dst + off), 0, 0);
    }
    wait_asynccnt0();
  }
#else
  {
    const v16h* src = (const v16h*)(Hh + (size_t)row0 * 512);
    v16h* dst = (v16h*)At;
#pragma unroll
    for (int it = 0; it < 8; ++it) dst[it * 256 + tid] = src[it * 256 + tid];
  }
#endif
  __syncthreads();
  int wave = tid >> 5, lane = tid & 31;
  int mi = wave & 3;                 // 4 M-tiles of 16 rows
  int nj = wave >> 2;                // 2 N-groups of 64 cols
  int mrow = mi * 16;
  int n0 = nBase + nj * 64;
  int r = lane & 15, hi = lane >> 4;
  v8f acc0 = {}, acc1 = {}, acc2 = {}, acc3 = {};
  for (int k0 = 0; k0 < 512; k0 += 32) {
    const h16* pa = At + (mrow + r) * 512 + k0 + hi * 16;
    V16 a; a.h[0] = *(const v8h*)pa; a.h[1] = *(const v8h*)(pa + 8);
    const h16* pb = Bt + (size_t)(n0 + r) * 512 + k0 + hi * 16;
    v16h b0 = *(const v16h*)pb;
    v16h b1 = *(const v16h*)(pb + 16 * 512);
    v16h b2 = *(const v16h*)(pb + 32 * 512);
    v16h b3 = *(const v16h*)(pb + 48 * 512);
    acc0 = __builtin_amdgcn_wmma_f32_16x16x32_f16(false, a.v, false, b0, (short)0, acc0, false, false);
    acc1 = __builtin_amdgcn_wmma_f32_16x16x32_f16(false, a.v, false, b1, (short)0, acc1, false, false);
    acc2 = __builtin_amdgcn_wmma_f32_16x16x32_f16(false, a.v, false, b2, (short)0, acc2, false, false);
    acc3 = __builtin_amdgcn_wmma_f32_16x16x32_f16(false, a.v, false, b3, (short)0, acc3, false, false);
  }
#pragma unroll
  for (int i = 0; i < 8; i++) {
    int rr = row0 + mrow + i + hi * 8;
    float* orow = Out + (size_t)rr * V_;
    int c0 = n0 + r;
    orow[c0]      = acc0[i] + bias[c0];
    orow[c0 + 16] = acc1[i] + bias[c0 + 16];
    orow[c0 + 32] = acc2[i] + bias[c0 + 32];
    orow[c0 + 48] = acc3[i] + bias[c0 + 48];
  }
}

// ---------------------------------------------------------------------------
// In-place log_softmax over each 32000-wide row (online max/sum, then write)
__global__ void k_logsoftmax(float* __restrict__ Out) {
  __shared__ float sm[256], ss[256];
  int r = blockIdx.x, tid = threadIdx.x;
  float* p = Out + (size_t)r * V_;
  float m = -3.0e38f, s = 0.f;
  for (int i = tid; i < V_; i += 256) {
    float x = p[i];
    if (x > m) { s = s * __expf(m - x) + 1.f; m = x; }
    else       { s += __expf(x - m); }
  }
  sm[tid] = m; ss[tid] = s;
  __syncthreads();
  for (int st = 128; st > 0; st >>= 1) {
    if (tid < st) {
      float m2 = sm[tid + st], s2 = ss[tid + st];
      float M = fmaxf(sm[tid], m2);
      ss[tid] = ss[tid] * __expf(sm[tid] - M) + s2 * __expf(m2 - M);
      sm[tid] = M;
    }
    __syncthreads();
  }
  float lse = sm[0] + __logf(ss[0]);
  for (int i = tid; i < V_; i += 256) p[i] -= lse;
}

// Final h, c copies into d_out tail
__global__ void k_final(const float* __restrict__ h_state, const float* __restrict__ c_state,
                        float* __restrict__ Out) {
  int idx = blockIdx.x * blockDim.x + threadIdx.x;   // 32768
  const size_t base = (size_t)B_ * L_ * V_;
  Out[base + idx] = h_state[idx];
  Out[base + 32768 + idx] = c_state[idx];
}

// ---------------------------------------------------------------------------
extern "C" void kernel_launch(void* const* d_in, const int* in_sizes, int n_in,
                              void* d_out, int out_size, void* d_ws, size_t ws_size,
                              hipStream_t stream) {
  (void)in_sizes; (void)n_in; (void)out_size; (void)ws_size;
  const float* features    = (const float*)d_in[0];
  const int*   topics      = (const int*)  d_in[1];
  const int*   captions    = (const int*)  d_in[2];
  const float* W_h_vis     = (const float*)d_in[3];
  const float* W_c_vis     = (const float*)d_in[4];
  const float* W_h_topic   = (const float*)d_in[5];
  const float* W_c_topic   = (const float*)d_in[6];
  const float* attn_vw_W   = (const float*)d_in[7];
  const float* topic_embed = (const float*)d_in[13];
  const float* word_embed  = (const float*)d_in[14];
  const float* W_ih        = (const float*)d_in[15];
  const float* W_hh        = (const float*)d_in[16];
  const float* b_ih        = (const float*)d_in[17];
  const float* b_hh        = (const float*)d_in[18];
  const float* out_W       = (const float*)d_in[19];
  const float* out_b       = (const float*)d_in[20];
  float* out = (float*)d_out;

  // workspace carve (256B aligned)
  char* ws = (char*)d_ws;
  size_t off = 0;
  auto carve = [&](size_t bytes) -> char* {
    char* p = ws + off;
    off = (off + bytes + 255) & ~(size_t)255;
    return p;
  };
  float* h_state = (float*)carve((size_t)B_ * H_ * 4);
  float* c_state = (float*)carve((size_t)B_ * H_ * 4);
  float* gates   = (float*)carve((size_t)B_ * G4_ * 4);
  h16*   Xcat    = (h16*)  carve((size_t)B_ * KX_ * 2);
  h16*   Wcat    = (h16*)  carve((size_t)G4_ * KX_ * 2);
  h16*   outWt   = (h16*)  carve((size_t)V_ * 512 * 2);
  h16*   Hall    = (h16*)  carve((size_t)B_ * L_ * 512 * 2);

  k_cvt_wcat<<<(G4_ * KX_) / 256, 256, 0, stream>>>(W_ih, W_hh, Wcat);
  k_cvt_outw<<<dim3(V_ / 32, 512 / 32), dim3(32, 8), 0, stream>>>(out_W, outWt);
  k_init<<<B_, 256, 0, stream>>>(features, topics, captions,
                                 W_h_vis, W_c_vis, W_h_topic, W_c_topic,
                                 attn_vw_W, topic_embed, word_embed,
                                 h_state, c_state, Xcat);
  for (int t = 0; t < L_; t++) {
    k_gates<<<64, 256, 0, stream>>>(Xcat, Wcat, gates);
    k_lstm<<<(B_ * H_) / 256, 256, 0, stream>>>(gates, b_ih, b_hh, h_state, c_state,
                                                Hall, Xcat, word_embed, captions, t);
  }
  k_outgemm<<<dim3((B_ * L_) / 64, V_ / 128), 256, 0, stream>>>(Hall, outWt, out_b, out);
  k_logsoftmax<<<B_ * L_, 256, 0, stream>>>(out);
  k_final<<<(B_ * H_) / 256, 256, 0, stream>>>(h_state, c_state, out);
}
